// Decoder_6055903887927
// MI455X (gfx1250) — compile-verified, matrix operand order checked by reading, not run on hardware
//
#include <hip/hip_runtime.h>
#include <cstdint>
#include <cstddef>

// ---------------------------------------------------------------------------
// Problem constants (match reference): A=128 atoms, T=2048, F=128, RBF=16,
// irrep dims {1,3,5}, grid N=65536. Output fp32 (N,F).
// ---------------------------------------------------------------------------
#define T_PTS 2048
#define F_DIM 128

typedef __bf16 bf16_t;
typedef __attribute__((ext_vector_type(16))) __bf16 v16bf;
typedef __attribute__((ext_vector_type(8)))  float  v8f;

__device__ __forceinline__ v8f wmma_bf16(v16bf a, v16bf b, v8f c) {
  // D(16x16,f32) = A(16x32,bf16) * B(32x16,bf16) + C
  return __builtin_amdgcn_wmma_f32_16x16x32_bf16(false, a, false, b,
                                                 (short)0, c, false, false);
}

// Fast SiLU: x * v_rcp_f32(1 + exp(-x)); bf16 pipeline tolerates approx rcp.
__device__ __forceinline__ float silu_fast(float x) {
  return x * __builtin_amdgcn_rcpf(1.0f + __expf(-x));
}

// Load a 16x32 bf16 fragment (A-layout; B uses same layout on a transposed,
// row-major-per-output-column matrix). base = &M[row0][0], ld in elements
// (multiple of 8 so each 8-elem run is 16B aligned).
// Per ISA 7.12.2: lane L<16 -> row L, K {0..7} and {16..23};
//                 lane L>=16 -> row L-16, K {8..15} and {24..31}.
__device__ __forceinline__ v16bf ldfrag(const bf16_t* base, int ld, int lane,
                                        int k0) {
  const bf16_t* p = base + (size_t)(lane & 15) * ld + k0 + ((lane >> 4) << 3);
  union { v16bf v; uint4 u[2]; } f;
  f.u[0] = *(const uint4*)(p);
  f.u[1] = *(const uint4*)(p + 16);
  return f.v;
}

// ---------------------------------------------------------------------------
// Kernel 1: transpose MLP weights to bf16, "output-column major" so WMMA
// B-fragments are two contiguous 16B loads per lane.
// w1t/w2t: [3][128][128], w3t: [3][2048][128]
// ---------------------------------------------------------------------------
__global__ __launch_bounds__(256) void prep_weights_kernel(
    const float* __restrict__ w1, const float* __restrict__ w2,
    const float* __restrict__ w3, bf16_t* __restrict__ w1t,
    bf16_t* __restrict__ w2t, bf16_t* __restrict__ w3t) {
  int g = blockIdx.x * 256 + threadIdx.x;  // grid sized exactly: 884736
  if (g < 49152) {                         // w1: 3*128*128
    int i = g >> 14, rem = g & 16383;
    int j = rem >> 7, k = rem & 127;       // dest [i][j][k] = src [i][k][j]
    w1t[g] = (bf16_t)w1[i * 16384 + k * 128 + j];
  } else if (g < 98304) {                  // w2
    int l = g - 49152;
    int i = l >> 14, rem = l & 16383;
    int j = rem >> 7, k = rem & 127;
    w2t[l] = (bf16_t)w2[i * 16384 + k * 128 + j];
  } else {                                 // w3: 3*128*2048 -> [i][2048][128]
    int l = g - 98304;
    int i = l / 262144, rem = l - i * 262144;
    int j = rem >> 7, k = rem & 127;
    w3t[l] = (bf16_t)w3[(size_t)i * 262144 + (size_t)k * 2048 + j];
  }
}

// ---------------------------------------------------------------------------
// Kernel 2: 3-layer MLP with WMMA.  grid = (3 irreps, 8 col-chunks), 256 thr.
// inv = feat0 (128x128 fp32).  Wout[i][n][r*128+f] fp32 (includes 1/sqrt(16)).
// ---------------------------------------------------------------------------
__global__ __launch_bounds__(256) void mlp_kernel(
    const float* __restrict__ feat0, const float* __restrict__ b1,
    const float* __restrict__ b2, const float* __restrict__ b3,
    const bf16_t* __restrict__ w1t, const bf16_t* __restrict__ w2t,
    const bf16_t* __restrict__ w3t, float* __restrict__ Wout) {
  __shared__ __align__(16) bf16_t Ha[128][136];  // row stride 272B (16B mult)
  __shared__ __align__(16) bf16_t Hb[128][136];
  const int i = blockIdx.x;
  const int nchunk = blockIdx.y;  // 256 output cols of layer-3 per block
  const int tid = threadIdx.x, lane = tid & 31, wid = tid >> 5;

  for (int e = tid; e < 128 * 128; e += 256)
    Ha[e >> 7][e & 127] = (bf16_t)feat0[e];
  __syncthreads();

  const bf16_t* W1 = w1t + i * 16384;
  const bf16_t* W2 = w2t + i * 16384;
  const bf16_t* W3 = w3t + (size_t)i * 262144;
  const float* B1 = b1 + i * 128;
  const float* B2 = b2 + i * 128;
  const float* B3 = b3 + i * 2048;
  float* Wo = Wout + (size_t)i * 262144;

  // ---- layer 1: Hb = silu(Ha @ W1 + B1), 64 16x16 tiles over 8 waves ----
  for (int tix = 0; tix < 8; ++tix) {
    int tile = wid * 8 + tix;
    int mt = tile >> 3, nt = tile & 7;
    v8f acc;
#pragma unroll
    for (int q = 0; q < 8; ++q) acc[q] = 0.0f;
#pragma unroll
    for (int ks = 0; ks < 4; ++ks) {
      v16bf a = ldfrag(&Ha[mt * 16][0], 136, lane, ks * 32);
      v16bf b = ldfrag(W1 + nt * 16 * 128, 128, lane, ks * 32);
      acc = wmma_bf16(a, b, acc);
    }
    int col = nt * 16 + (lane & 15);
    int rbase = mt * 16 + ((lane >> 4) << 3);
#pragma unroll
    for (int v = 0; v < 8; ++v)
      Hb[rbase + v][col] = (bf16_t)silu_fast(acc[v] + B1[col]);
  }
  __syncthreads();

  // ---- layer 2: Ha = silu(Hb @ W2 + B2) ----
  for (int tix = 0; tix < 8; ++tix) {
    int tile = wid * 8 + tix;
    int mt = tile >> 3, nt = tile & 7;
    v8f acc;
#pragma unroll
    for (int q = 0; q < 8; ++q) acc[q] = 0.0f;
#pragma unroll
    for (int ks = 0; ks < 4; ++ks) {
      v16bf a = ldfrag(&Hb[mt * 16][0], 136, lane, ks * 32);
      v16bf b = ldfrag(W2 + nt * 16 * 128, 128, lane, ks * 32);
      acc = wmma_bf16(a, b, acc);
    }
    int col = nt * 16 + (lane & 15);
    int rbase = mt * 16 + ((lane >> 4) << 3);
#pragma unroll
    for (int v = 0; v < 8; ++v)
      Ha[rbase + v][col] = (bf16_t)silu_fast(acc[v] + B2[col]);
  }
  __syncthreads();

  // ---- layer 3: Wo[:, nchunk*256 : +256] = (Ha @ W3 + B3) * 0.25 ----
  for (int tix = 0; tix < 16; ++tix) {  // 8 mt x 16 nt tiles over 8 waves
    int tile = wid * 16 + tix;
    int mt = tile >> 4, nt = tile & 15;
    int jn = nchunk * 256 + nt * 16;
    v8f acc;
#pragma unroll
    for (int q = 0; q < 8; ++q) acc[q] = 0.0f;
#pragma unroll
    for (int ks = 0; ks < 4; ++ks) {
      v16bf a = ldfrag(&Ha[mt * 16][0], 136, lane, ks * 32);
      v16bf b = ldfrag(W3 + (size_t)jn * 128, 128, lane, ks * 32);
      acc = wmma_bf16(a, b, acc);
    }
    int col = jn + (lane & 15);
    int rbase = mt * 16 + ((lane >> 4) << 3);
#pragma unroll
    for (int v = 0; v < 8; ++v)
      Wo[(size_t)(rbase + v) * 2048 + col] = (acc[v] + B3[col]) * 0.25f;
  }
}

// ---------------------------------------------------------------------------
// Kernel 3: Y[(n,d,r), f] = feat[n,f,d] * W[n,r,f], stored f-major bf16
// (Yt[f*K + k]) so GEMM B-fragments are contiguous.  One thread per element.
// ---------------------------------------------------------------------------
template <int DIM>
__global__ __launch_bounds__(256) void build_y_kernel(
    const float* __restrict__ feat, const float* __restrict__ W,
    bf16_t* __restrict__ Yt) {
  constexpr int K = 2048 * DIM;
  int g = blockIdx.x * 256 + threadIdx.x;  // grid sized exactly: K*128/256
  int f = g / K;
  int k = g - f * K;
  int n = k / (16 * DIM);
  int rem = k - n * (16 * DIM);
  int d = rem >> 4, r = rem & 15;
  float gv = feat[(size_t)n * (128 * DIM) + f * DIM + d];
  float wv = W[(size_t)n * 2048 + r * 128 + f];
  Yt[g] = (bf16_t)(gv * wv);
}

// ---------------------------------------------------------------------------
// Kernel 4: main GEMM  sparse[t,f] = sum_k X[t,k] * Y[k,f]
//   X[t,k] built on the fly into LDS bf16: sph[n,t,d]*rbf[n,t,r]
//   Double-buffered X tile: fill chunk c+1 overlaps WMMA on chunk c,
//   one barrier per 64-K chunk.
//   grid = (T/64 t-blocks, 4 K-slices); block = 256 (8 waves, 4x2 wave grid)
//   each wave: 16(t) x 64(f) via 4 accumulators; 8 WMMA per 64-K chunk.
// ---------------------------------------------------------------------------
template <int DIM>
__device__ __forceinline__ void fill_x(bf16_t (&X)[64][72],
                                       const float* __restrict__ S,
                                       const float* __restrict__ R, int t,
                                       int tl, int j0, int k0) {
  constexpr int D16 = 16 * DIM;
  int k = k0 + j0;
  int n = k / D16;
  int rem = k - n * D16;
#pragma unroll
  for (int j = 0; j < 16; ++j) {
    int d = rem >> 4, r = rem & 15;
    float sv = S[(size_t)n * (T_PTS * DIM) + (size_t)t * DIM + d];
    float rv = R[(size_t)n * (T_PTS * 16) + (size_t)t * 16 + r];
    X[tl][j0 + j] = (bf16_t)(sv * rv);
    if (++rem == D16) { rem = 0; ++n; }
  }
}

template <int DIM>
__global__ __launch_bounds__(256) void sparse_gemm_kernel(
    const float* __restrict__ S, const float* __restrict__ R,
    const bf16_t* __restrict__ Yt, float* __restrict__ sp) {
  constexpr int K = 2048 * DIM;
  constexpr int KS = K / 4;       // K per slice (multiple of 64)
  constexpr int NCH = KS / 64;    // chunks per slice
  __shared__ __align__(16) bf16_t Xs[2][64][72];  // row stride 144B
  const int tid = threadIdx.x, lane = tid & 31, wid = tid >> 5;
  const int wm = wid & 3, wn = wid >> 2;
  const int t0 = blockIdx.x * 64;
  const int kbeg = blockIdx.y * KS;
  const int tl = tid >> 2;            // 0..63 : t row of X this thread fills
  const int j0 = (tid & 3) << 4;      // 0,16,32,48 : k column base
  const int t = t0 + tl;

  // per-wave B row base pointers (f-major Y, ld = K)
  const bf16_t* Brow[4];
#pragma unroll
  for (int nt = 0; nt < 4; ++nt)
    Brow[nt] = Yt + (size_t)(wn * 64 + nt * 16) * K;

  v8f acc[4];
#pragma unroll
  for (int nt = 0; nt < 4; ++nt)
#pragma unroll
    for (int q = 0; q < 8; ++q) acc[nt][q] = 0.0f;

  fill_x<DIM>(Xs[0], S, R, t, tl, j0, kbeg);
  __syncthreads();

  for (int c = 0; c < NCH; ++c) {
    const int k0 = kbeg + c * 64;
    // overlap: build next chunk's X while issuing WMMA on current chunk
    if (c + 1 < NCH) fill_x<DIM>(Xs[(c + 1) & 1], S, R, t, tl, j0, k0 + 64);
#pragma unroll
    for (int ks = 0; ks < 2; ++ks) {
      v16bf a = ldfrag(&Xs[c & 1][wm * 16][0], 72, lane, ks * 32);
#pragma unroll
      for (int nt = 0; nt < 4; ++nt) {
        v16bf b = ldfrag(Brow[nt], K, lane, k0 + ks * 32);
        acc[nt] = wmma_bf16(a, b, acc[nt]);
        // prefetch next chunk's B region (global_prefetch_b8)
        if (c + 1 < NCH)
          __builtin_prefetch(Brow[nt] + (size_t)(lane & 15) * K + k0 + 64 +
                                 ks * 32,
                             0, 0);
      }
    }
    __syncthreads();
  }

  // store partial tile into this slice's slab
  float* out = sp + (size_t)blockIdx.y * (T_PTS * F_DIM);
  const int rbase = wm * 16 + ((lane >> 4) << 3);
#pragma unroll
  for (int nt = 0; nt < 4; ++nt) {
    int col = wn * 64 + nt * 16 + (lane & 15);
#pragma unroll
    for (int v = 0; v < 8; ++v)
      out[(size_t)(t0 + rbase + v) * F_DIM + col] = acc[nt][v];
  }
}

// ---------------------------------------------------------------------------
// Kernel 5: sum the 12 partial slabs (3 irreps x 4 K-slices) and scatter-add
// into the (N,F) grid. truncated_idx may contain duplicates -> atomicAdd.
// ---------------------------------------------------------------------------
__global__ __launch_bounds__(256) void scatter_kernel(
    const float* __restrict__ sparse, const int* __restrict__ idx,
    float* __restrict__ out) {
  int g = blockIdx.x * 256 + threadIdx.x;  // < T*F = 262144
  int tt = g >> 7, f = g & 127;
  float s = 0.0f;
#pragma unroll
  for (int slab = 0; slab < 12; ++slab)
    s += sparse[(size_t)slab * (T_PTS * F_DIM) + g];
  atomicAdd(&out[(size_t)idx[tt] * F_DIM + f], s);
}

// ---------------------------------------------------------------------------
// Launch
// ---------------------------------------------------------------------------
extern "C" void kernel_launch(void* const* d_in, const int* in_sizes, int n_in,
                              void* d_out, int out_size, void* d_ws,
                              size_t ws_size, hipStream_t stream) {
  const float* feat0 = (const float*)d_in[0];
  const float* feat1 = (const float*)d_in[1];
  const float* feat2 = (const float*)d_in[2];
  const float* sph0  = (const float*)d_in[3];
  const float* sph1  = (const float*)d_in[4];
  const float* sph2  = (const float*)d_in[5];
  const float* rbf   = (const float*)d_in[6];
  const float* w1    = (const float*)d_in[7];
  const float* b1    = (const float*)d_in[8];
  const float* w2    = (const float*)d_in[9];
  const float* b2    = (const float*)d_in[10];
  const float* w3    = (const float*)d_in[11];
  const float* b3    = (const float*)d_in[12];
  const int*   idx   = (const int*)d_in[13];

  // workspace layout (all offsets 16B aligned), total ~22.2 MB:
  char* ws = (char*)d_ws;
  float*  Wbuf   = (float*)(ws + 0);          // 3*128*2048 f32  = 3,145,728 B
  bf16_t* w1t    = (bf16_t*)(ws + 3145728);   // 3*128*128 bf16  =    98,304 B
  bf16_t* w2t    = (bf16_t*)(ws + 3244032);   //                      98,304 B
  bf16_t* w3t    = (bf16_t*)(ws + 3342336);   // 3*2048*128 bf16 = 1,572,864 B
  bf16_t* Yt     = (bf16_t*)(ws + 4915200);   // 18432*128 bf16  = 4,718,592 B
  float*  sparse = (float*)(ws + 9633792);    // 12*2048*128 f32 =12,582,912 B

  hipMemsetAsync(d_out, 0, (size_t)out_size * sizeof(float), stream);

  prep_weights_kernel<<<3456, 256, 0, stream>>>(w1, w2, w3, w1t, w2t, w3t);
  mlp_kernel<<<dim3(3, 8), 256, 0, stream>>>(feat0, b1, b2, b3, w1t, w2t, w3t,
                                             Wbuf);

  build_y_kernel<1><<<1024, 256, 0, stream>>>(feat0, Wbuf,          Yt);
  build_y_kernel<3><<<3072, 256, 0, stream>>>(feat1, Wbuf + 262144, Yt + 262144);
  build_y_kernel<5><<<5120, 256, 0, stream>>>(feat2, Wbuf + 524288, Yt + 1048576);

  sparse_gemm_kernel<1><<<dim3(32, 4), 256, 0, stream>>>(sph0, rbf, Yt,
                                                         sparse);
  sparse_gemm_kernel<3><<<dim3(32, 4), 256, 0, stream>>>(sph1, rbf,
                                                         Yt + 262144,
                                                         sparse + 4 * 262144);
  sparse_gemm_kernel<5><<<dim3(32, 4), 256, 0, stream>>>(sph2, rbf,
                                                         Yt + 1048576,
                                                         sparse + 8 * 262144);

  scatter_kernel<<<1024, 256, 0, stream>>>(sparse, idx, (float*)d_out);
}
